// SelfAttentionLayer_22557168239379
// MI455X (gfx1250) — compile-verified, matrix operand order checked by reading, not run on hardware
//
#include <hip/hip_runtime.h>

typedef __bf16 bf16;
typedef __attribute__((ext_vector_type(16))) __bf16 v16bf;
typedef __attribute__((ext_vector_type(8)))  __bf16 v8bf;
typedef __attribute__((ext_vector_type(8)))  float  v8f;
typedef __attribute__((ext_vector_type(4)))  int    v4i;

// ---------------------------------------------------------------------------
// CDNA5 async global->LDS copy (ASYNCcnt-tracked, bypasses VGPRs).
// Builtin signature (from toolchain diagnostic): 
//   void(v4i addrspace(1)* src, v4i addrspace(3)* dst, imm offset, imm cpol)
// Per ISA 10/§15.18, INST_OFFSET applies to BOTH the LDS and the global
// address, so one (lds,global) base pair serves several 16B beats.
// ---------------------------------------------------------------------------
#if defined(__has_builtin)
#if __has_builtin(__builtin_amdgcn_global_load_async_to_lds_b128)
#define HAVE_ASYNC_LDS 1
#endif
#endif

#ifdef HAVE_ASYNC_LDS
#define ASYNC_B128(ldsdst, gsrc, imm)                                          \
  __builtin_amdgcn_global_load_async_to_lds_b128(                              \
      (__attribute__((address_space(1))) v4i*)(uintptr_t)(gsrc),               \
      (__attribute__((address_space(3))) v4i*)(uintptr_t)(ldsdst), (imm), 0)
#if __has_builtin(__builtin_amdgcn_s_wait_asynccnt)
#define ASYNC_WAIT() __builtin_amdgcn_s_wait_asynccnt(0)
#else
#define ASYNC_WAIT() asm volatile("s_wait_asynccnt 0" ::: "memory")
#endif
#else
#define ASYNC_B128(ldsdst, gsrc, imm)                                          \
  (*(v8bf*)((char*)(ldsdst) + (imm)) =                                         \
       *(const v8bf*)((const char*)(gsrc) + (imm)))
#define ASYNC_WAIT() ((void)0)
#endif

// ---------------------------------------------------------------------------
// Load one 16-row x 32-col (16-bit) WMMA A/B fragment from row-major storage.
// Per CDNA5 ISA 16-bit operand layout: lane L -> row (L&15); lanes 0-15 hold
// K in {0..7, 16..23}, lanes 16-31 hold K in {8..15, 24..31}; v16bf elements
// 0..7 = first 8-chunk, 8..15 = second 8-chunk.
// ---------------------------------------------------------------------------
__device__ __forceinline__ v16bf load_frag16(const bf16* base, int ld, int lane) {
  const int row = lane & 15;
  const int c0  = (lane >> 4) * 8;   // 0 or 8
  const bf16* p = base + row * ld + c0;
  v8bf lo = *(const v8bf*)(p);
  v8bf hi = *(const v8bf*)(p + 16);
  v16bf a;
#pragma unroll
  for (int e = 0; e < 8; ++e) { a[e] = lo[e]; a[e + 8] = hi[e]; }
  return a;
}

__device__ __forceinline__ v8f wmma_bf16(v16bf a, v16bf b, v8f c) {
  return __builtin_amdgcn_wmma_f32_16x16x32_bf16(false, a, false, b, (short)0, c,
                                                 false, false);
}

// ---------------------------------------------------------------------------
// fp32 -> bf16 conversion
// ---------------------------------------------------------------------------
__global__ void f32_to_bf16_kernel(const float* __restrict__ src,
                                   bf16* __restrict__ dst, int n) {
  int i = blockIdx.x * blockDim.x + threadIdx.x;
  if (i < n) dst[i] = (bf16)src[i];
}

// ---------------------------------------------------------------------------
// QKV GEMM (out = X @ Wqkv^T) with fused RMS-norm epilogue for q/k sections.
// X: [4096,1024] bf16, W: [3072,1024] bf16.  Tile: 128 rows x 64 cols,
// K staged in 64-deep slabs, double-buffered via async global->LDS copies.
// Grid: x = 48 col-tiles, y = 32 row-tiles.  256 threads (8 waves).
// ---------------------------------------------------------------------------
__global__ __launch_bounds__(256) void qkv_gemm_norm_kernel(
    const bf16* __restrict__ xb, const bf16* __restrict__ wq,
    const float* __restrict__ q_scale, const float* __restrict__ k_scale,
    bf16* __restrict__ qb, bf16* __restrict__ kb, bf16* __restrict__ vb) {
  const int C = 1024, Nq = 2048, H = 16;
  __shared__ bf16 Xs[2][128][72];   // 64 K-slab + 8 pad (row stride 144B)
  __shared__ bf16 Ws[2][64][72];
  const int tid = threadIdx.x, lane = tid & 31, wave = tid >> 5;
  const int row0 = blockIdx.y * 128;
  const int col0 = blockIdx.x * 64;

  auto stage = [&](int buf, int kk) {
    { // X tile 128x64: each thread one 32-element row chunk (4 x b128)
      int r = tid >> 1, c = (tid & 1) * 32;
      const bf16* g = xb + (size_t)(row0 + r) * C + kk + c;
      bf16* d = &Xs[buf][r][c];
      ASYNC_B128(d, g, 0);  ASYNC_B128(d, g, 16);
      ASYNC_B128(d, g, 32); ASYNC_B128(d, g, 48);
    }
    { // W tile 64x64: each thread one 16-element chunk (2 x b128)
      int r = tid >> 2, c = (tid & 3) * 16;
      const bf16* g = wq + (size_t)(col0 + r) * C + kk + c;
      bf16* d = &Ws[buf][r][c];
      ASYNC_B128(d, g, 0);  ASYNC_B128(d, g, 16);
    }
  };

  v8f acc[4] = {};
  stage(0, 0);
  for (int kk = 0; kk < C; kk += 64) {
    const int cur = (kk >> 6) & 1;
    ASYNC_WAIT();
    __syncthreads();                       // cur slab ready; prev compute done
    if (kk + 64 < C) stage(cur ^ 1, kk + 64);   // overlap fetch with compute
    v16bf a0 = load_frag16(&Xs[cur][wave * 16][0], 72, lane);
    v16bf a1 = load_frag16(&Xs[cur][wave * 16][32], 72, lane);
#pragma unroll
    for (int t = 0; t < 4; ++t) {
      v16bf b0 = load_frag16(&Ws[cur][t * 16][0], 72, lane);
      v16bf b1 = load_frag16(&Ws[cur][t * 16][32], 72, lane);
      acc[t] = wmma_bf16(a0, b0, acc[t]);
      acc[t] = wmma_bf16(a1, b1, acc[t]);
    }
  }

  // Epilogue. C/D layout: lanes 0-15 rows 0-7, lanes 16-31 rows 8-15; col=lane&15.
  const int half = lane >> 4, nloc = lane & 15;
  const int sec  = col0 / C;            // 0=q 1=k 2=v
  const int head = (col0 % C) >> 6;

  if (sec < 2) {
    const float* sc = (sec == 0) ? q_scale : k_scale;
    bf16* dst = (sec == 0) ? qb : kb;
#pragma unroll
    for (int r = 0; r < 8; ++r) {
      float ss = 0.f;
#pragma unroll
      for (int t = 0; t < 4; ++t) { float v = acc[t][r]; ss += v * v; }
      ss += __shfl_xor(ss, 1, 32);
      ss += __shfl_xor(ss, 2, 32);
      ss += __shfl_xor(ss, 4, 32);
      ss += __shfl_xor(ss, 8, 32);
      float rms = sqrtf(ss) * 0.125f;          // ||x|| / sqrt(64)
      float inv = 1.0f / (rms + 1e-8f);
      int gm = row0 + wave * 16 + r + 8 * half;     // 0..4095
      int b  = gm >> 11, n = gm & 2047;
      size_t base = (((size_t)b * H + head) * Nq + n) * 64;
#pragma unroll
      for (int t = 0; t < 4; ++t) {
        int colh = nloc + 16 * t;
        dst[base + colh] = (bf16)(acc[t][r] * inv * sc[colh]);
      }
    }
  } else {
#pragma unroll
    for (int r = 0; r < 8; ++r) {
      int gm = row0 + wave * 16 + r + 8 * half;
      int b  = gm >> 11, n = gm & 2047;
      size_t base = (((size_t)b * H + head) * Nq + n) * 64;
#pragma unroll
      for (int t = 0; t < 4; ++t)
        vb[base + nloc + 16 * t] = (bf16)acc[t][r];
    }
  }
}

// ---------------------------------------------------------------------------
// Flash attention.  Grid: x = N/128 query blocks, y = B*H.  256 thr (8 waves).
// Wave owns 16 query rows; iterates over 64-key blocks with online softmax.
// K block staged asynchronously, overlapped with the V-transpose VALU work.
// ---------------------------------------------------------------------------
__global__ __launch_bounds__(256) void flash_attn_kernel(
    const bf16* __restrict__ qb, const bf16* __restrict__ kb,
    const bf16* __restrict__ vb, bf16* __restrict__ ao) {
  const int Nq = 2048, HD = 64, C = 1024;
  __shared__ bf16 Ks[64][72];        // K rows, row-major  (9216 B)
  __shared__ bf16 Vt[64][72];        // V transposed: [col][key] (9216 B)
  __shared__ bf16 Pw[8][16][72];     // per-wave P staging (18432 B)

  const int tid = threadIdx.x, lane = tid & 31, wave = tid >> 5;
  const int half = lane >> 4, nloc = lane & 15;
  const int bh = blockIdx.y, b = bh >> 4, h = bh & 15;
  const int qrow0 = blockIdx.x * 128 + wave * 16;
  const size_t headbase = (size_t)bh * Nq * HD;

  // Q fragments (16x64 -> two 16x32 A-frags), kept in registers.
  const bf16* qbase = qb + headbase + (size_t)qrow0 * HD;
  v16bf qa0 = load_frag16(qbase, HD, lane);
  v16bf qa1 = load_frag16(qbase + 32, HD, lane);

  float mrow[8], lrow[8];
#pragma unroll
  for (int r = 0; r < 8; ++r) { mrow[r] = -1e30f; lrow[r] = 0.f; }
  v8f acco[4] = {};

  for (int j = 0; j < Nq; j += 64) {
    __syncthreads();   // protect Ks/Vt from previous iteration's readers
    {
      int r = tid >> 2, c = (tid & 3) * 16;
      // K rows: async global->LDS (overlaps with the V transpose below)
      const bf16* gk = kb + headbase + (size_t)(j + r) * HD + c;
      ASYNC_B128(&Ks[r][c], gk, 0);
      ASYNC_B128(&Ks[r][c], gk, 16);
      // V transposed through VGPRs
      const bf16* gv = vb + headbase + (size_t)(j + r) * HD + c;
      v8bf y0 = *(const v8bf*)gv;
      v8bf y1 = *(const v8bf*)(gv + 8);
#pragma unroll
      for (int e = 0; e < 8; ++e) {
        Vt[c + e][r]     = y0[e];
        Vt[c + 8 + e][r] = y1[e];
      }
    }
    ASYNC_WAIT();
    __syncthreads();

    // S = Q @ K^T  (16x64 per wave, 8 WMMAs)
    v8f s[4] = {};
#pragma unroll
    for (int t = 0; t < 4; ++t) {
      v16bf bk0 = load_frag16(&Ks[t * 16][0], 72, lane);
      v16bf bk1 = load_frag16(&Ks[t * 16][32], 72, lane);
      s[t] = wmma_bf16(qa0, bk0, s[t]);
      s[t] = wmma_bf16(qa1, bk1, s[t]);
    }

    // online softmax update; stage P (bf16) to LDS
#pragma unroll
    for (int r = 0; r < 8; ++r) {
      float mx = -1e30f;
#pragma unroll
      for (int t = 0; t < 4; ++t) mx = fmaxf(mx, s[t][r] * 0.125f);
      mx = fmaxf(mx, __shfl_xor(mx, 1, 32));
      mx = fmaxf(mx, __shfl_xor(mx, 2, 32));
      mx = fmaxf(mx, __shfl_xor(mx, 4, 32));
      mx = fmaxf(mx, __shfl_xor(mx, 8, 32));
      float mnew = fmaxf(mrow[r], mx);
      float corr = __expf(mrow[r] - mnew);
      float psum = 0.f;
      int m = r + 8 * half;
#pragma unroll
      for (int t = 0; t < 4; ++t) {
        float p = __expf(s[t][r] * 0.125f - mnew);
        psum += p;
        Pw[wave][m][nloc + 16 * t] = (bf16)p;
        acco[t][r] = acco[t][r] * corr;
      }
      psum += __shfl_xor(psum, 1, 32);
      psum += __shfl_xor(psum, 2, 32);
      psum += __shfl_xor(psum, 4, 32);
      psum += __shfl_xor(psum, 8, 32);
      lrow[r] = lrow[r] * corr + psum;
      mrow[r] = mnew;
    }
    __syncthreads();   // P visible (and Vt still intact)

    // O += P @ V  (8 WMMAs)
    v16bf pa0 = load_frag16(&Pw[wave][0][0], 72, lane);
    v16bf pa1 = load_frag16(&Pw[wave][0][32], 72, lane);
#pragma unroll
    for (int t = 0; t < 4; ++t) {
      v16bf bv0 = load_frag16(&Vt[t * 16][0], 72, lane);
      v16bf bv1 = load_frag16(&Vt[t * 16][32], 72, lane);
      acco[t] = wmma_bf16(pa0, bv0, acco[t]);
      acco[t] = wmma_bf16(pa1, bv1, acco[t]);
    }
  }

  // normalize and store to [B,N,C] bf16 for the proj GEMM
#pragma unroll
  for (int r = 0; r < 8; ++r) {
    float inv = 1.0f / lrow[r];
    int n = qrow0 + r + 8 * half;
    size_t base = ((size_t)b * Nq + n) * C + h * 64;
#pragma unroll
    for (int t = 0; t < 4; ++t)
      ao[base + nloc + 16 * t] = (bf16)(acco[t][r] * inv);
  }
}

// ---------------------------------------------------------------------------
// Output projection: out = AO @ Wp^T + bias (fp32 out).  Same pipelined
// double-buffered 64-deep K slab as the QKV GEMM.  Tile 128x64.
// Grid: x = 16 col-tiles, y = 32 row-tiles.
// ---------------------------------------------------------------------------
__global__ __launch_bounds__(256) void proj_gemm_kernel(
    const bf16* __restrict__ ab, const bf16* __restrict__ wp,
    const float* __restrict__ bias, float* __restrict__ out) {
  const int C = 1024;
  __shared__ bf16 Xs[2][128][72];
  __shared__ bf16 Ws[2][64][72];
  const int tid = threadIdx.x, lane = tid & 31, wave = tid >> 5;
  const int row0 = blockIdx.y * 128;
  const int col0 = blockIdx.x * 64;

  auto stage = [&](int buf, int kk) {
    {
      int r = tid >> 1, c = (tid & 1) * 32;
      const bf16* g = ab + (size_t)(row0 + r) * C + kk + c;
      bf16* d = &Xs[buf][r][c];
      ASYNC_B128(d, g, 0);  ASYNC_B128(d, g, 16);
      ASYNC_B128(d, g, 32); ASYNC_B128(d, g, 48);
    }
    {
      int r = tid >> 2, c = (tid & 3) * 16;
      const bf16* g = wp + (size_t)(col0 + r) * C + kk + c;
      bf16* d = &Ws[buf][r][c];
      ASYNC_B128(d, g, 0);  ASYNC_B128(d, g, 16);
    }
  };

  v8f acc[4] = {};
  stage(0, 0);
  for (int kk = 0; kk < C; kk += 64) {
    const int cur = (kk >> 6) & 1;
    ASYNC_WAIT();
    __syncthreads();
    if (kk + 64 < C) stage(cur ^ 1, kk + 64);
    v16bf a0 = load_frag16(&Xs[cur][wave * 16][0], 72, lane);
    v16bf a1 = load_frag16(&Xs[cur][wave * 16][32], 72, lane);
#pragma unroll
    for (int t = 0; t < 4; ++t) {
      v16bf b0 = load_frag16(&Ws[cur][t * 16][0], 72, lane);
      v16bf b1 = load_frag16(&Ws[cur][t * 16][32], 72, lane);
      acc[t] = wmma_bf16(a0, b0, acc[t]);
      acc[t] = wmma_bf16(a1, b1, acc[t]);
    }
  }

  const int half = lane >> 4, nloc = lane & 15;
#pragma unroll
  for (int r = 0; r < 8; ++r) {
    int gm = row0 + wave * 16 + r + 8 * half;
#pragma unroll
    for (int t = 0; t < 4; ++t) {
      int col = col0 + nloc + 16 * t;
      out[(size_t)gm * C + col] = acc[t][r] + bias[col];
    }
  }
}

// ---------------------------------------------------------------------------
extern "C" void kernel_launch(void* const* d_in, const int* in_sizes, int n_in,
                              void* d_out, int out_size, void* d_ws, size_t ws_size,
                              hipStream_t stream) {
  const float* x       = (const float*)d_in[0];   // [2,2048,1024]
  const float* qkv_w   = (const float*)d_in[1];   // [3072,1024]
  const float* proj_w  = (const float*)d_in[2];   // [1024,1024]
  const float* proj_b  = (const float*)d_in[3];   // [1024]
  const float* q_scale = (const float*)d_in[4];   // [64]
  const float* k_scale = (const float*)d_in[5];   // [64]
  float* out = (float*)d_out;

  char* ws = (char*)d_ws;
  bf16* xb    = (bf16*)(ws);                        //  8 MB  x bf16
  bf16* wqkv  = (bf16*)(ws + ( 8u << 20));          //  6 MB
  bf16* wproj = (bf16*)(ws + (14u << 20));          //  2 MB
  bf16* qb    = (bf16*)(ws + (16u << 20));          //  8 MB  [B,H,N,64]
  bf16* kb    = (bf16*)(ws + (24u << 20));          //  8 MB
  bf16* vb    = (bf16*)(ws + (32u << 20));          //  8 MB
  bf16* ao    = (bf16*)(ws + (40u << 20));          //  8 MB  [B,N,C]

  const int nx = 2 * 2048 * 1024;      // 4194304
  const int nw = 3072 * 1024;          // 3145728
  const int np = 1024 * 1024;          // 1048576
  f32_to_bf16_kernel<<<(nx + 255) / 256, 256, 0, stream>>>(x, xb, nx);
  f32_to_bf16_kernel<<<(nw + 255) / 256, 256, 0, stream>>>(qkv_w, wqkv, nw);
  f32_to_bf16_kernel<<<(np + 255) / 256, 256, 0, stream>>>(proj_w, wproj, np);

  qkv_gemm_norm_kernel<<<dim3(48, 32), 256, 0, stream>>>(
      xb, wqkv, q_scale, k_scale, qb, kb, vb);

  flash_attn_kernel<<<dim3(16, 32), 256, 0, stream>>>(qb, kb, vb, ao);

  proj_gemm_kernel<<<dim3(16, 32), 256, 0, stream>>>(ao, wproj, proj_b, out);
}
